// TutorialTransformer_14285061226680
// MI455X (gfx1250) — compile-verified, matrix operand order checked by reading, not run on hardware
//
#include <hip/hip_runtime.h>
#include <hip/hip_bf16.h>
#include <math.h>

typedef __attribute__((ext_vector_type(16))) _Float16 v16h;
typedef __attribute__((ext_vector_type(8)))  float    v8f;

#define C0F 0.28209479177387814f
#define LN_EPS 1e-5f

struct RadialParams {
    const float *W1, *b1, *g1, *be1;
    const float *W2, *b2, *g2, *be2;
    const float *W3, *b3;
};

// ---- order-preserving float<->uint encode for atomicMax-based segment max ----
__device__ __forceinline__ unsigned fenc(float x) {
    unsigned u = __float_as_uint(x);
    return (u & 0x80000000u) ? ~u : (u | 0x80000000u);
}
__device__ __forceinline__ float fdec(unsigned u) {
    return __uint_as_float((u & 0x80000000u) ? (u & 0x7FFFFFFFu) : ~u);
}

// A/B fragment K index mapping for v_wmma_f32_16x16x32_f16 (16-bit, 16x32):
// lanes 0-15 hold K={0..7,16..23}, lanes 16-31 hold K={8..15,24..31}.
__device__ __forceinline__ int kmap(int i, int lane) {
    return ((lane & 16) ? 8 : 0) + (i < 8 ? i : i + 8);
}

// Build B fragment (KxN tile, N tile index `ntile`) from fp32 row-major W[k*stride+col].
__device__ __forceinline__ v16h bfrag(const float* W, int stride, int kvalid,
                                      int ntile, int nvalid, int lane) {
    int col = ntile * 16 + (lane & 15);
    v16h b;
#pragma unroll
    for (int i = 0; i < 16; ++i) {
        int k = kmap(i, lane);
        float v = 0.0f;
        if (k < kvalid && col < nvalid) v = W[k * stride + col];
        b[i] = (_Float16)v;
    }
    return b;
}

// Build A fragment from LDS staging tile with LayerNorm (stats in cols 32/33) + ReLU.
__device__ __forceinline__ v16h afrag_ln(const float* S, int lane,
                                         const float* g, const float* be) {
    int m = lane & 15;
    float mu  = S[m * 36 + 32];
    float inv = S[m * 36 + 33];
    v16h a;
#pragma unroll
    for (int i = 0; i < 16; ++i) {
        int k = kmap(i, lane);
        float xv = (S[m * 36 + k] - mu) * inv * g[k] + be[k];
        a[i] = (_Float16)fmaxf(xv, 0.0f);
    }
    return a;
}

// Store two 16x16 f32 accumulator tiles (+per-feature bias) to LDS staging tile.
// Accumulator layout: lane = column (within tile), VGPR r = row r (lanes 0-15) / r+8 (16-31).
__device__ __forceinline__ void store_pair(float* S, int lane, v8f d0, v8f d1,
                                           const float* bias) {
    int c = lane & 15;
    int ro = (lane & 16) ? 8 : 0;
    float b0 = bias[c], b1 = bias[16 + c];
#pragma unroll
    for (int r = 0; r < 8; ++r) {
        S[(ro + r) * 36 + c]      = d0[r] + b0;
        S[(ro + r) * 36 + 16 + c] = d1[r] + b1;
    }
}

// Store final single tile (+bias, *C0) to LDS.
__device__ __forceinline__ void store_final(float* S, int lane, v8f d,
                                            const float* b3, int nvalid) {
    int c = lane & 15;
    int ro = (lane & 16) ? 8 : 0;
    float bb = (c < nvalid) ? b3[c] : 0.0f;
#pragma unroll
    for (int r = 0; r < 8; ++r)
        S[(ro + r) * 36 + c] = (d[r] + bb) * C0F;
}

// Per-row LayerNorm stats over 32 features; lanes 0-15 each handle one row.
__device__ __forceinline__ void ln_stats(float* S, int lane) {
    if (lane < 16) {
        float s = 0.0f;
#pragma unroll
        for (int k = 0; k < 32; ++k) s += S[lane * 36 + k];
        float mu = s * (1.0f / 32.0f);
        float v = 0.0f;
#pragma unroll
        for (int k = 0; k < 32; ++k) {
            float d = S[lane * 36 + k] - mu;
            v += d * d;
        }
        v *= (1.0f / 32.0f);
        S[lane * 36 + 32] = mu;
        S[lane * 36 + 33] = rsqrtf(v + LN_EPS);
    }
}

#define WMMA(Afrag, Bfrag, Cacc) \
    __builtin_amdgcn_wmma_f32_16x16x32_f16(false, (Afrag), false, (Bfrag), (short)0, (Cacc), false, false)

// Full radial MLP (Linear->LN->ReLU->Linear->LN->ReLU->Linear) for a wave's
// 16-edge tile; final (C0-scaled, biased) kernel matrix left in LDS tile S.
__device__ void radial_net(const float feat[5], int lane, float* S,
                           const RadialParams p, int nout) {
    __syncthreads();                       // protect previous epilogue's reads of S
    v8f zc = {};
    // ---- layer 1: [16x5(pad32)] x [32x32] ----
    v16h a;
#pragma unroll
    for (int i = 0; i < 16; ++i) {
        int k = kmap(i, lane);
        a[i] = (_Float16)((k < 5) ? feat[k] : 0.0f);
    }
    v8f d0 = WMMA(a, bfrag(p.W1, 32, 5, 0, 32, lane), zc);
    v8f d1 = WMMA(a, bfrag(p.W1, 32, 5, 1, 32, lane), zc);
    store_pair(S, lane, d0, d1, p.b1);
    __syncthreads();
    ln_stats(S, lane);
    __syncthreads();
    // ---- layer 2: [16x32] x [32x32] ----
    a = afrag_ln(S, lane, p.g1, p.be1);
    __syncthreads();                       // reads done before overwrite
    d0 = WMMA(a, bfrag(p.W2, 32, 32, 0, 32, lane), zc);
    d1 = WMMA(a, bfrag(p.W2, 32, 32, 1, 32, lane), zc);
    store_pair(S, lane, d0, d1, p.b2);
    __syncthreads();
    ln_stats(S, lane);
    __syncthreads();
    // ---- layer 3: [16x32] x [32 x nout(pad16)] ----
    a = afrag_ln(S, lane, p.g2, p.be2);
    __syncthreads();
    d0 = WMMA(a, bfrag(p.W3, nout, 32, 0, nout, lane), zc);
    store_final(S, lane, d0, p.b3, nout);
    __syncthreads();                       // make final tile visible to epilogue
}

// =============== Kernel 1: q = f0 @ Wq^T ===============
__global__ void node_q_kernel(const float* __restrict__ f, const float* __restrict__ Wq,
                              float* __restrict__ q, int N) {
    int n = blockIdx.x * blockDim.x + threadIdx.x;
    if (n >= N) return;
    float f0 = f[n * 4 + 0], f1 = f[n * 4 + 1], f2 = f[n * 4 + 2], f3 = f[n * 4 + 3];
#pragma unroll
    for (int o = 0; o < 3; ++o)
        q[n * 3 + o] = f0 * Wq[o * 4 + 0] + f1 * Wq[o * 4 + 1] +
                       f2 * Wq[o * 4 + 2] + f3 * Wq[o * 4 + 3];
}

// =============== Kernel 2: fused WMMA edge MLPs ===============
__global__ void __launch_bounds__(256)
edge_mlp_kernel(const float* __restrict__ f, const float* __restrict__ x,
                const float* __restrict__ w, const int* __restrict__ src,
                const int* __restrict__ dst,
                RadialParams pv, RadialParams pk, RadialParams po,
                const float* __restrict__ qbuf,
                float* __restrict__ vbuf, float* __restrict__ escbuf,
                float* __restrict__ kobuf,
                unsigned* __restrict__ emaxEnc, float* __restrict__ deg,
                int E) {
    __shared__ float Sall[8 * 16 * 36];
    int lane = threadIdx.x & 31;
    int wv   = threadIdx.x >> 5;
    float* S = Sall + wv * 16 * 36;
    long long eb = ((long long)blockIdx.x * 8 + wv) * 16;
    int m = lane & 15;
    long long e = eb + m;
    bool ev = (e < (long long)E);

    float feat[5] = {0.f, 0.f, 0.f, 0.f, 0.f};
    if (ev) {
        const float4 w4 = ((const float4*)w)[e];
        int s = src[e], d = dst[e];
        float dx = x[d * 3 + 0] - x[s * 3 + 0];
        float dy = x[d * 3 + 1] - x[s * 3 + 1];
        float dz = x[d * 3 + 2] - x[s * 3 + 2];
        feat[0] = w4.x; feat[1] = w4.y; feat[2] = w4.z; feat[3] = w4.w;
        feat[4] = sqrtf(dx * dx + dy * dy + dz * dz);
    }

    // ---- value kernel: v = Kv @ f0[src] ----
    radial_net(feat, lane, S, pv, 12);
    if (lane < 16 && ev) {
        int s = src[e];
        float fs0 = f[s * 4 + 0], fs1 = f[s * 4 + 1], fs2 = f[s * 4 + 2], fs3 = f[s * 4 + 3];
#pragma unroll
        for (int o = 0; o < 3; ++o) {
            float val = S[lane * 36 + o * 4 + 0] * fs0 + S[lane * 36 + o * 4 + 1] * fs1 +
                        S[lane * 36 + o * 4 + 2] * fs2 + S[lane * 36 + o * 4 + 3] * fs3;
            vbuf[e * 3 + o] = val;
        }
    }

    // ---- key kernel: score = (Kk@f0[src]) . q[dst] / sqrt(3); segment max + degree ----
    radial_net(feat, lane, S, pk, 12);
    if (lane < 16 && ev) {
        int s = src[e], d = dst[e];
        float fs0 = f[s * 4 + 0], fs1 = f[s * 4 + 1], fs2 = f[s * 4 + 2], fs3 = f[s * 4 + 3];
        float esc = 0.0f;
#pragma unroll
        for (int o = 0; o < 3; ++o) {
            float kk = S[lane * 36 + o * 4 + 0] * fs0 + S[lane * 36 + o * 4 + 1] * fs1 +
                       S[lane * 36 + o * 4 + 2] * fs2 + S[lane * 36 + o * 4 + 3] * fs3;
            esc += kk * qbuf[d * 3 + o];
        }
        esc *= 0.57735026919f;  // 1/sqrt(M_MID=3)
        escbuf[e] = esc;
        atomicMax(&emaxEnc[d], fenc(esc));
        atomicAdd(&deg[d], 1.0f);
    }

    // ---- output-conv kernel: store Ko (3x3, C0-scaled) per edge ----
    radial_net(feat, lane, S, po, 9);
    if (lane < 16 && ev) {
#pragma unroll
        for (int j = 0; j < 9; ++j) kobuf[e * 9 + j] = S[lane * 36 + j];
    }
}

// =============== Kernel 3: ez = exp(e - max), denom accumulation ===============
__global__ void edge_ez_kernel(const float* __restrict__ esc, const int* __restrict__ dst,
                               const unsigned* __restrict__ emaxEnc,
                               float* __restrict__ ez, float* __restrict__ denom, int E) {
    int e = blockIdx.x * blockDim.x + threadIdx.x;
    if (e >= E) return;
    int d = dst[e];
    float z = expf(esc[e] - fdec(emaxEnc[d]));
    ez[e] = z;
    atomicAdd(&denom[d], z);
}

// =============== Kernel 4: attn[dst] += (ez/denom[dst]) * v ===============
__global__ void edge_attn_kernel(const float* __restrict__ ez, const float* __restrict__ denom,
                                 const float* __restrict__ vbuf, const int* __restrict__ dst,
                                 float* __restrict__ attn, int E) {
    int e = blockIdx.x * blockDim.x + threadIdx.x;
    if (e >= E) return;
    int d = dst[e];
    float a = ez[e] / denom[d];
#pragma unroll
    for (int o = 0; o < 3; ++o)
        atomicAdd(&attn[d * 3 + o], a * vbuf[e * 3 + o]);
}

// =============== Kernel 5: skip-cat projection + GNormSE3 ===============
__global__ void node_zn_kernel(const float* __restrict__ attn, const float* __restrict__ f,
                               const float* __restrict__ Wp, const float* __restrict__ n_g,
                               const float* __restrict__ n_b, const float* __restrict__ n_W,
                               const float* __restrict__ n_bW, float* __restrict__ zn, int N) {
    int n = blockIdx.x * blockDim.x + threadIdx.x;
    if (n >= N) return;
    float z[3], nz[3], ph[3];
#pragma unroll
    for (int o = 0; o < 3; ++o) {
        float acc = 0.0f;
#pragma unroll
        for (int j = 0; j < 3; ++j) acc += attn[n * 3 + j] * Wp[o * 7 + j];
#pragma unroll
        for (int i = 0; i < 4; ++i) acc += f[n * 4 + i] * Wp[o * 7 + 3 + i];
        z[o] = acc;
        nz[o] = fmaxf(fabsf(acc), 1e-12f);
        ph[o] = z[o] / nz[o];
    }
    float mu = (nz[0] + nz[1] + nz[2]) * (1.0f / 3.0f);
    float var = 0.0f;
#pragma unroll
    for (int o = 0; o < 3; ++o) { float d = nz[o] - mu; var += d * d; }
    var *= (1.0f / 3.0f);
    float inv = rsqrtf(var + LN_EPS);
    float t0[3];
#pragma unroll
    for (int o = 0; o < 3; ++o)
        t0[o] = fmaxf((nz[o] - mu) * inv * n_g[o] + n_b[o], 0.0f);
#pragma unroll
    for (int o = 0; o < 3; ++o) {
        float t = n_bW[o];
#pragma unroll
        for (int j = 0; j < 3; ++j) t += t0[j] * n_W[o * 3 + j];
        zn[n * 3 + o] = t * ph[o];
    }
}

// =============== Kernel 6: acc[dst] += Ko @ zn[src] ===============
__global__ void edge_msg_kernel(const float* __restrict__ kobuf, const float* __restrict__ zn,
                                const int* __restrict__ src, const int* __restrict__ dst,
                                float* __restrict__ acc, int E) {
    int e = blockIdx.x * blockDim.x + threadIdx.x;
    if (e >= E) return;
    int s = src[e], d = dst[e];
    float z0 = zn[s * 3 + 0], z1 = zn[s * 3 + 1], z2 = zn[s * 3 + 2];
#pragma unroll
    for (int o = 0; o < 3; ++o) {
        float m = kobuf[(long long)e * 9 + o * 3 + 0] * z0 +
                  kobuf[(long long)e * 9 + o * 3 + 1] * z1 +
                  kobuf[(long long)e * 9 + o * 3 + 2] * z2;
        atomicAdd(&acc[d * 3 + o], m);
    }
}

// =============== Kernel 7: out = acc/deg + zn@Ws^T (deg>0) ===============
__global__ void node_out_kernel(const float* __restrict__ acc, const float* __restrict__ deg,
                                const float* __restrict__ zn, const float* __restrict__ Ws,
                                float* __restrict__ out, int N) {
    int n = blockIdx.x * blockDim.x + threadIdx.x;
    if (n >= N) return;
    float dg = deg[n];
#pragma unroll
    for (int o = 0; o < 3; ++o) {
        float r = 0.0f;
        if (dg > 0.0f) {
            r = acc[n * 3 + o] / dg;
#pragma unroll
            for (int j = 0; j < 3; ++j) r += zn[n * 3 + j] * Ws[o * 3 + j];
        }
        out[n * 3 + o] = r;
    }
}

extern "C" void kernel_launch(void* const* d_in, const int* in_sizes, int n_in,
                              void* d_out, int out_size, void* d_ws, size_t ws_size,
                              hipStream_t stream) {
    const float* f   = (const float*)d_in[0];
    const float* x   = (const float*)d_in[1];
    const float* w   = (const float*)d_in[2];
    const int*   src = (const int*)d_in[3];
    const int*   dst = (const int*)d_in[4];
    const int N = in_sizes[0] / 4;
    const int E = in_sizes[3];

    auto FP = [&](int i) { return (const float*)d_in[i]; };
    RadialParams pv = {FP(5),  FP(6),  FP(7),  FP(8),  FP(9),  FP(10), FP(11), FP(12), FP(13), FP(14)};
    RadialParams pk = {FP(15), FP(16), FP(17), FP(18), FP(19), FP(20), FP(21), FP(22), FP(23), FP(24)};
    RadialParams po = {FP(25), FP(26), FP(27), FP(28), FP(29), FP(30), FP(31), FP(32), FP(33), FP(34)};
    const float* Wq  = FP(35);
    const float* Wp  = FP(36);
    const float* n_g = FP(37);
    const float* n_b = FP(38);
    const float* n_W = FP(39);
    const float* n_bW = FP(40);
    const float* Ws  = FP(41);

    // ---- workspace layout ----
    char* base = (char*)d_ws;
    size_t off = 0;
    auto take = [&](size_t elems) {
        void* p = base + off;
        off = (off + elems * sizeof(float) + 255) & ~(size_t)255;
        return p;
    };
    // zeroed node-accumulator region (contiguous, one memset)
    size_t zstart = off;
    unsigned* emaxEnc = (unsigned*)take(N);   // encoded segment max
    float* denom = (float*)take(N);
    float* attn  = (float*)take(3 * (size_t)N);
    float* deg   = (float*)take(N);
    float* acc   = (float*)take(3 * (size_t)N);
    size_t zbytes = off - zstart;
    // non-zeroed buffers
    float* qbuf  = (float*)take(3 * (size_t)N);
    float* zn    = (float*)take(3 * (size_t)N);
    float* vbuf  = (float*)take(3 * (size_t)E);
    float* esc   = (float*)take((size_t)E);
    float* ez    = (float*)take((size_t)E);
    float* kobuf = (float*)take(9 * (size_t)E);
    (void)ws_size; (void)n_in;

    float* out = (float*)d_out; (void)out_size;

    hipMemsetAsync(base + zstart, 0, zbytes, stream);

    const int TB = 256;
    node_q_kernel<<<(N + TB - 1) / TB, TB, 0, stream>>>(f, Wq, qbuf, N);

    int tiles = (E + 15) / 16;
    int blocks = (tiles + 7) / 8;      // 8 waves (tiles) per 256-thread block
    edge_mlp_kernel<<<blocks, TB, 0, stream>>>(f, x, w, src, dst, pv, pk, po,
                                               qbuf, vbuf, esc, kobuf, emaxEnc, deg, E);

    edge_ez_kernel<<<(E + TB - 1) / TB, TB, 0, stream>>>(esc, dst, emaxEnc, ez, denom, E);
    edge_attn_kernel<<<(E + TB - 1) / TB, TB, 0, stream>>>(ez, denom, vbuf, dst, attn, E);
    node_zn_kernel<<<(N + TB - 1) / TB, TB, 0, stream>>>(attn, f, Wp, n_g, n_b, n_W, n_bW, zn, N);
    edge_msg_kernel<<<(E + TB - 1) / TB, TB, 0, stream>>>(kobuf, zn, src, dst, acc, E);
    node_out_kernel<<<(N + TB - 1) / TB, TB, 0, stream>>>(acc, deg, zn, Ws, out, N);
}
